// PretrainedTokenizer_50079318671613
// MI455X (gfx1250) — compile-verified, matrix operand order checked by reading, not run on hardware
//
#include <hip/hip_runtime.h>
#include <hip/hip_bf16.h>

typedef __attribute__((ext_vector_type(16))) _Float16 v16h;
typedef __attribute__((ext_vector_type(8)))  float    v8f;

#define WAVES 8
#define NB 2  // batch

__device__ __forceinline__ unsigned lds_off(const void* p) {
  // generic pointer into LDS: low 32 bits are the LDS byte address
  return (unsigned)(unsigned long long)p;
}

// ---------------------------------------------------------------------------
// Async implicit-GEMM conv (SAME, stride 1) on V_WMMA_F32_16X16X32_F16.
// Requires: f16 input, Cin%32==0, Cout%32==0.  One wave owns 16 pixels x 32
// Cout.  A (16x32) and B (32x32) f16 tiles are fetched with
// GLOBAL_LOAD_ASYNC_TO_LDS_B128 into double-buffered wave-private LDS;
// s_wait_asynccnt 6 overlaps tile k+1 loads with tile k WMMAs.  Halo pixels
// are redirected to a zero pad buffer so the async copy performs zero-fill.
// ---------------------------------------------------------------------------
template<int KS>
__global__ __launch_bounds__(256) void conv_wmma_async_kernel(
    const _Float16* __restrict__ in, const _Float16* __restrict__ wgt,
    const float* __restrict__ bias, const float* __restrict__ res,
    float* __restrict__ out, const _Float16* __restrict__ zpad,
    int lW, int lHW, int Cin, int Cout, int lcb)
{
  __shared__ _Float16 lA[2][WAVES][16][32];
  __shared__ _Float16 lB[2][WAVES][32][32];

  const int lane = threadIdx.x & 31;
  const int wv   = threadIdx.x >> 5;
  const int half = lane >> 4;
  const int l16  = lane & 15;

  const int W  = 1 << lW;
  const int H  = 1 << (lHW - lW);
  const int HW = 1 << lHW;

  const int Mtiles = (NB << lHW) >> 4;
  int mt = blockIdx.x * WAVES + wv;
  const bool active = (mt < Mtiles);
  if (!active) mt = 0;
  const int mbase = mt << 4;
  const int nbase = blockIdx.y << 5;
  constexpr int pad = KS >> 1;

  const int pA   = mbase + l16;
  const int nimg = pA >> lHW;
  const int rem  = pA & (HW - 1);
  const int ph   = rem >> lW;
  const int pw   = rem & (W - 1);

  const int ksteps = (KS * KS) << lcb;

  const unsigned offA0 = lds_off(&lA[0][wv][l16][half << 4]);
  const unsigned offA1 = lds_off(&lA[1][wv][l16][half << 4]);
  const unsigned offB0 = lds_off(&lB[0][wv][lane][0]);
  const unsigned offB1 = lds_off(&lB[1][wv][lane][0]);

  auto issue = [&](int kt) {
    const int bufi = kt & 1;
    const int kk   = kt & ((1 << lcb) - 1);
    const int pos  = kt >> lcb;
    const int ky   = pos / KS, kx = pos - ky * KS;
    const int sh   = ph + ky - pad, sw = pw + kx - pad;
    const _Float16* ga =
        (sh >= 0 && sh < H && sw >= 0 && sw < W)
            ? in + ((size_t)(nimg * HW + (sh << lW) + sw)) * Cin + (kk << 5) + (half << 4)
            : zpad;
    const _Float16* gb = wgt + ((size_t)((kt << 5) + lane)) * Cout + nbase;
    const unsigned oa = bufi ? offA1 : offA0;
    const unsigned ob = bufi ? offB1 : offB0;
    asm volatile(
        "global_load_async_to_lds_b128 %0, %2, off\n\t"
        "global_load_async_to_lds_b128 %0, %2, off offset:16\n\t"
        "global_load_async_to_lds_b128 %1, %3, off\n\t"
        "global_load_async_to_lds_b128 %1, %3, off offset:16\n\t"
        "global_load_async_to_lds_b128 %1, %3, off offset:32\n\t"
        "global_load_async_to_lds_b128 %1, %3, off offset:48"
        :: "v"(oa), "v"(ob), "v"(ga), "v"(gb)
        : "memory");
  };

  v8f acc[2];
  acc[0] = (v8f){}; acc[1] = (v8f){};

  issue(0);
  for (int kt = 0; kt < ksteps; ++kt) {
    if (kt + 1 < ksteps) {
      issue(kt + 1);
      asm volatile("s_wait_asynccnt 0x6" ::: "memory");
    } else {
      asm volatile("s_wait_asynccnt 0x0" ::: "memory");
    }
    const int bi = kt & 1;
    // gather per ISA 16-bit A/B layouts: e<8 -> K=e+8h ; e>=8 -> K=16+(e-8)+8h
    v16h a;
    #pragma unroll
    for (int e = 0; e < 16; ++e) {
      int kk = ((e & 8) << 1) + (e & 7) + (half << 3);
      a[e] = lA[bi][wv][l16][kk];
    }
    #pragma unroll
    for (int j = 0; j < 2; ++j) {
      v16h b;
      #pragma unroll
      for (int e = 0; e < 16; ++e) {
        int kk = ((e & 8) << 1) + (e & 7) + (half << 3);
        b[e] = lB[bi][wv][kk][(j << 4) + l16];
      }
      acc[j] = __builtin_amdgcn_wmma_f32_16x16x32_f16(
          false, a, false, b, (short)0, acc[j], false, false);
    }
  }

  if (active) {
    #pragma unroll
    for (int j = 0; j < 2; ++j) {
      const int co = nbase + (j << 4) + l16;
      const float bv = bias ? bias[co] : 0.f;
      #pragma unroll
      for (int r = 0; r < 8; ++r) {
        size_t off = (size_t)(mbase + r + (half << 3)) * Cout + co;
        float v = acc[j][r] + bv;
        if (res) v += res[off];
        out[off] = v;
      }
    }
  }
}

// ---------------------------------------------------------------------------
// Synchronous conv for the irregular shapes (Cin=3 conv_in, Cout=3 final,
// f32-input nin 1x1).  Weights are f16; input f32 or f16 per IN16.
// ---------------------------------------------------------------------------
template<int KS, int NT, bool AFAST, bool BFAST, bool IN16>
__global__ __launch_bounds__(256) void conv_wmma_sync_kernel(
    const void* __restrict__ inv, const _Float16* __restrict__ wgt,
    const float* __restrict__ bias, const float* __restrict__ res,
    float* __restrict__ out,
    int lW, int lHW, int Cin, int Cout, int lcb, int doClip)
{
  constexpr int NW = NT * 16;
  __shared__ _Float16 lA[WAVES][16][32];
  __shared__ _Float16 lB[WAVES][32][NW];

  const float*    inF = (const float*)inv;
  const _Float16* inH = (const _Float16*)inv;

  const int lane = threadIdx.x & 31;
  const int wv   = threadIdx.x >> 5;
  const int half = lane >> 4;
  const int l16  = lane & 15;

  const int W  = 1 << lW;
  const int H  = 1 << (lHW - lW);
  const int HW = 1 << lHW;

  const int Mtiles = (NB << lHW) >> 4;
  int mt = blockIdx.x * WAVES + wv;
  const bool active = (mt < Mtiles);
  if (!active) mt = 0;
  const int mbase = mt << 4;
  const int nbase = blockIdx.y * NW;
  constexpr int pad = KS >> 1;

  const int pA   = mbase + l16;
  const int nimg = pA >> lHW;
  const int rem  = pA & (HW - 1);
  const int ph   = rem >> lW;
  const int pw   = rem & (W - 1);

  const int Ktot   = KS * KS * Cin;
  const int ksteps = AFAST ? (KS * KS) << lcb : (Ktot + 31) >> 5;

  v8f acc[NT];
  #pragma unroll
  for (int j = 0; j < NT; ++j) acc[j] = (v8f){};

  for (int kt = 0; kt < ksteps; ++kt) {
    // ---------------- stage A ----------------
    if (AFAST) {
      const int kk  = kt & ((1 << lcb) - 1);
      const int pos = kt >> lcb;
      const int ky  = pos / KS, kx = pos - ky * KS;
      const int sh  = ph + ky - pad, sw = pw + kx - pad;
      const int cb  = (kk << 5) + (half << 4);
      if (sh >= 0 && sh < H && sw >= 0 && sw < W) {
        const size_t base = ((size_t)(nimg * HW + (sh << lW) + sw)) * Cin + cb;
        if (IN16) {
          const uint4* s = (const uint4*)(inH + base);
          *(uint4*)&lA[wv][l16][(half << 4) + 0] = s[0];
          *(uint4*)&lA[wv][l16][(half << 4) + 8] = s[1];
        } else {
          #pragma unroll
          for (int i = 0; i < 16; i += 4) {
            float4 v = *(const float4*)(inF + base + i);
            lA[wv][l16][(half << 4) + i + 0] = (_Float16)v.x;
            lA[wv][l16][(half << 4) + i + 1] = (_Float16)v.y;
            lA[wv][l16][(half << 4) + i + 2] = (_Float16)v.z;
            lA[wv][l16][(half << 4) + i + 3] = (_Float16)v.w;
          }
        }
      } else {
        uint4 z = {0, 0, 0, 0};
        *(uint4*)&lA[wv][l16][(half << 4) + 0] = z;
        *(uint4*)&lA[wv][l16][(half << 4) + 8] = z;
      }
    } else {
      #pragma unroll
      for (int i = 0; i < 16; ++i) {
        int e  = (lane << 4) + i;
        int m  = e >> 5, kl = e & 31;
        int kg = (kt << 5) + kl;
        _Float16 v = (_Float16)0.f;
        if (kg < Ktot) {
          int c   = kg % Cin;
          int pos = kg / Cin;
          int ky  = pos / KS, kx = pos - ky * KS;
          int pp  = mbase + m;
          int ni  = pp >> lHW;
          int rr  = pp & (HW - 1);
          int hh  = rr >> lW, ww = rr & (W - 1);
          int sh  = hh + ky - pad, sw = ww + kx - pad;
          if (sh >= 0 && sh < H && sw >= 0 && sw < W)
            v = (_Float16)inF[((size_t)(ni * HW + (sh << lW) + sw)) * Cin + c];
        }
        lA[wv][m][kl] = v;
      }
    }
    // ---------------- stage B ----------------
    {
      const int kl = lane;
      const int kg = (kt << 5) + kl;
      if (kg < Ktot) {
        const _Float16* wsrc = wgt + (size_t)kg * Cout + nbase;
        if (BFAST) {
          #pragma unroll
          for (int i = 0; i < NW / 8; ++i)
            *(uint4*)&lB[wv][kl][i * 8] = ((const uint4*)wsrc)[i];
        } else {
          #pragma unroll
          for (int i = 0; i < NW; ++i)
            lB[wv][kl][i] = (nbase + i < Cout) ? wsrc[i] : (_Float16)0.f;
        }
      } else {
        #pragma unroll
        for (int i = 0; i < NW / 8; ++i) {
          uint4 z = {0, 0, 0, 0};
          *(uint4*)&lB[wv][kl][i * 8] = z;
        }
      }
    }
    __syncthreads();

    v16h a;
    #pragma unroll
    for (int e = 0; e < 16; ++e) {
      int kk = ((e & 8) << 1) + (e & 7) + (half << 3);
      a[e] = lA[wv][l16][kk];
    }
    #pragma unroll
    for (int j = 0; j < NT; ++j) {
      v16h b;
      #pragma unroll
      for (int e = 0; e < 16; ++e) {
        int kk = ((e & 8) << 1) + (e & 7) + (half << 3);
        b[e] = lB[wv][kk][j * 16 + l16];
      }
      acc[j] = __builtin_amdgcn_wmma_f32_16x16x32_f16(
          false, a, false, b, (short)0, acc[j], false, false);
    }
    __syncthreads();
  }

  if (active) {
    #pragma unroll
    for (int j = 0; j < NT; ++j) {
      const int co = nbase + j * 16 + l16;
      if (co < Cout) {
        const float bv = bias ? bias[co] : 0.f;
        #pragma unroll
        for (int r = 0; r < 8; ++r) {
          size_t off = (size_t)(mbase + r + (half << 3)) * Cout + co;
          float v = acc[j][r] + bv;
          if (res)    v += res[off];
          if (doClip) v = fminf(fmaxf(v, 0.f), 1.f);
          out[off] = v;
        }
      }
    }
  }
}

// ---------------------------------------------------------------------------
// Fused GroupNorm(32) + SiLU, fp32 in -> f16 out.  Block per (batch, group).
// ---------------------------------------------------------------------------
__global__ __launch_bounds__(256) void gn_silu_kernel(
    const float* __restrict__ x, const float* __restrict__ gamma,
    const float* __restrict__ beta, _Float16* __restrict__ y, int HW, int lcpg)
{
  const int n   = blockIdx.x >> 5;
  const int g   = blockIdx.x & 31;
  const int C   = 32 << lcpg;
  const float*    xb = x + (size_t)n * HW * C;
  _Float16*       yb = y + (size_t)n * HW * C;
  const int total = HW << lcpg;
  const int cbase = g << lcpg;

  float sum = 0.f, sq = 0.f;
  for (int i = threadIdx.x; i < total; i += 256) {
    int p = i >> lcpg, j = i & ((1 << lcpg) - 1);
    float v = xb[(size_t)p * C + cbase + j];
    sum += v; sq += v * v;
  }
  __shared__ float s1[256], s2[256];
  s1[threadIdx.x] = sum; s2[threadIdx.x] = sq;
  __syncthreads();
  for (int o = 128; o > 0; o >>= 1) {
    if (threadIdx.x < o) {
      s1[threadIdx.x] += s1[threadIdx.x + o];
      s2[threadIdx.x] += s2[threadIdx.x + o];
    }
    __syncthreads();
  }
  const float mu  = s1[0] / (float)total;
  const float var = s2[0] / (float)total - mu * mu;
  const float inv = rsqrtf(var + 1e-6f);

  for (int i = threadIdx.x; i < total; i += 256) {
    int p = i >> lcpg, j = i & ((1 << lcpg) - 1);
    int c = cbase + j;
    float v = xb[(size_t)p * C + c];
    float t = (v - mu) * inv * gamma[c] + beta[c];
    yb[(size_t)p * C + c] = (_Float16)(t / (1.f + __expf(-t)));   // SiLU
  }
}

// ---------------------------------------------------------------------------
// 2x2 average pool f32->f32; 2x nearest upsample f32->f16 (feeds conv).
// ---------------------------------------------------------------------------
__global__ void pool2_kernel(const float* __restrict__ in,
                             float* __restrict__ out, int lH2, int lW2, int lC)
{
  const size_t total = (size_t)NB << (lH2 + lW2 + lC);
  for (size_t i = (size_t)blockIdx.x * blockDim.x + threadIdx.x; i < total;
       i += (size_t)gridDim.x * blockDim.x) {
    int c  = (int)i & ((1 << lC) - 1);
    size_t r = i >> lC;
    int w2 = (int)r & ((1 << lW2) - 1); r >>= lW2;
    int h2 = (int)r & ((1 << lH2) - 1);
    int n  = (int)(r >> lH2);
    const size_t rowC = (size_t)1 << (lW2 + 1 + lC);   // W*C
    const size_t base =
        (((((size_t)n << (lH2 + 1)) + (h2 << 1)) << (lW2 + 1)) + (w2 << 1)) << lC;
    const float* b = in + base + c;
    out[i] = 0.25f * (b[0] + b[1 << lC] + b[rowC] + b[rowC + (1 << lC)]);
  }
}

__global__ void up2_kernel(const float* __restrict__ in,
                           _Float16* __restrict__ out, int lH, int lW, int lC)
{
  const size_t total = (size_t)NB << (lH + lW + 2 + lC);
  for (size_t i = (size_t)blockIdx.x * blockDim.x + threadIdx.x; i < total;
       i += (size_t)gridDim.x * blockDim.x) {
    int c = (int)i & ((1 << lC) - 1);
    size_t r = i >> lC;
    int w = (int)r & ((2 << lW) - 1); r >>= (lW + 1);
    int h = (int)r & ((2 << lH) - 1);
    int n = (int)(r >> (lH + 1));
    out[i] = (_Float16)in[((((size_t)n << lH) + (h >> 1)) << (lW + lC)) +
                          ((size_t)(w >> 1) << lC) + c];
  }
}

// ---------------------------------------------------------------------------
// fp32 -> f16 weight conversion; zero-fill for the async pad buffer.
// ---------------------------------------------------------------------------
__global__ void cvt16_kernel(const float* __restrict__ src,
                             _Float16* __restrict__ dst, int n)
{
  for (size_t i = (size_t)blockIdx.x * blockDim.x + threadIdx.x; i < (size_t)n;
       i += (size_t)gridDim.x * blockDim.x)
    dst[i] = (_Float16)src[i];
}

__global__ void zero16_kernel(_Float16* __restrict__ dst, int n)
{
  int i = blockIdx.x * blockDim.x + threadIdx.x;
  if (i < n) dst[i] = (_Float16)0.f;
}

// ---------------------------------------------------------------------------
// Vector quantization: nearest of 1024 codebook entries per 256-dim vector.
// fp32 in, f16 zq out (feeds decoder conv).
// ---------------------------------------------------------------------------
#define VQ_D  256
#define VQ_NE 1024
__global__ __launch_bounds__(256) void vq_kernel(
    const float* __restrict__ flat, const float* __restrict__ emb,
    _Float16* __restrict__ zq)
{
  const int v = blockIdx.x;
  __shared__ float fv[VQ_D];
  for (int c = threadIdx.x; c < VQ_D; c += 256) fv[c] = flat[(size_t)v * VQ_D + c];
  __syncthreads();

  float best = 3.4e38f; int bi = VQ_NE;
  for (int e = threadIdx.x; e < VQ_NE; e += 256) {
    const float* ev = emb + (size_t)e * VQ_D;
    float d = 0.f;
    #pragma unroll 4
    for (int c = 0; c < VQ_D; ++c) { float t = fv[c] - ev[c]; d += t * t; }
    if (d < best || (d == best && e < bi)) { best = d; bi = e; }
  }
  __shared__ float sb[256]; __shared__ int si[256];
  sb[threadIdx.x] = best; si[threadIdx.x] = bi;
  __syncthreads();
  for (int o = 128; o > 0; o >>= 1) {
    if (threadIdx.x < o) {
      if (sb[threadIdx.x + o] < sb[threadIdx.x] ||
          (sb[threadIdx.x + o] == sb[threadIdx.x] &&
           si[threadIdx.x + o] < si[threadIdx.x])) {
        sb[threadIdx.x] = sb[threadIdx.x + o];
        si[threadIdx.x] = si[threadIdx.x + o];
      }
    }
    __syncthreads();
  }
  const int idx = si[0];
  for (int c = threadIdx.x; c < VQ_D; c += 256)
    zq[(size_t)v * VQ_D + c] = (_Float16)emb[(size_t)idx * VQ_D + c];
}

// ---------------------------------------------------------------------------
// Host-side orchestration
// ---------------------------------------------------------------------------
static inline int lg2(int v) { int r = 0; while ((1 << r) < v) ++r; return r; }

extern "C" void kernel_launch(void* const* d_in, const int* in_sizes, int n_in,
                              void* d_out, int out_size, void* d_ws, size_t ws_size,
                              hipStream_t stream)
{
  (void)n_in; (void)out_size; (void)ws_size;

  // ---- replay jax tree-flatten order (dict keys sorted alphabetically) ----
  int t = 0;
  struct RB { int c1, c2, n1b, n1g, n2b, n2g, nin; };
  auto mk = [&](bool hasNin) {
    RB r; r.c1 = t++; r.c2 = t++; r.n1b = t++; r.n1g = t++;
    r.n2b = t++; r.n2g = t++; r.nin = hasNin ? t++ : -1; return r;
  };
  const int dec_cib = t++, dec_ciw = t++, dec_cob = t++, dec_cow = t++;
  RB dmid[2]; dmid[0] = mk(false); dmid[1] = mk(false);
  const int dec_nb = t++, dec_ng = t++;
  RB  up_b[5][2]; int up_ub[5], up_uw[5];
  up_b[0][0] = mk(false); up_b[0][1] = mk(false); up_ub[0] = -1;  up_uw[0] = -1;
  up_b[1][0] = mk(true);  up_b[1][1] = mk(false); up_ub[1] = t++; up_uw[1] = t++;
  up_b[2][0] = mk(false); up_b[2][1] = mk(false); up_ub[2] = t++; up_uw[2] = t++;
  up_b[3][0] = mk(true);  up_b[3][1] = mk(false); up_ub[3] = t++; up_uw[3] = t++;
  up_b[4][0] = mk(false); up_b[4][1] = mk(false); up_ub[4] = t++; up_uw[4] = t++;
  const int emb_i = t++;
  const int enc_cob = t++, enc_convin = t++, enc_cow = t++;
  RB down_b[5][2];
  down_b[0][0] = mk(false); down_b[0][1] = mk(false);
  down_b[1][0] = mk(false); down_b[1][1] = mk(false);
  down_b[2][0] = mk(true);  down_b[2][1] = mk(false);
  down_b[3][0] = mk(false); down_b[3][1] = mk(false);
  down_b[4][0] = mk(true);  down_b[4][1] = mk(false);
  RB emid[2]; emid[0] = mk(false); emid[1] = mk(false);
  const int enc_nb = t++, enc_ng = t++;
  const int x_i = t++;
  const int NT_TENS = t;

  auto P = [&](int i) -> const float* { return (const float*)d_in[i]; };

  // ---- workspace layout ----------------------------------------------------
  const size_t BUFE = (size_t)NB * 256 * 256 * 128;       // 16.7M elements
  float*    B0   = (float*)d_ws;                          // ping
  float*    B1   = B0 + BUFE;                             // pong
  float*    U32  = B1 + BUFE;                             // f32 temp
  _Float16* G16  = (_Float16*)(U32 + BUFE);               // f16 temp (gn out)
  _Float16* ZP   = G16 + BUFE;                            // zero pad (256 elems)
  _Float16* W16  = ZP + 256;                              // f16 weight pool

  // per-launch weight conversion bookkeeping (local state only)
  _Float16* wptr[256];
  bool      wdone[256];
  for (int i = 0; i < 256 && i < NT_TENS; ++i) wdone[i] = false;
  size_t wpos = 0;
  auto WH = [&](int wi) -> const _Float16* {
    if (!wdone[wi]) {
      wdone[wi] = true;
      wptr[wi] = W16 + wpos;
      int n = in_sizes[wi];
      wpos += ((size_t)n + 63) & ~(size_t)63;
      cvt16_kernel<<<dim3((n + 255) / 256), 256, 0, stream>>>(P(wi), wptr[wi], n);
    }
    return wptr[wi];
  };

  zero16_kernel<<<dim3(1), 256, 0, stream>>>(ZP, 256);

  auto gn = [&](const float* x, int gi, int bi, _Float16* y, int H, int W, int C) {
    gn_silu_kernel<<<dim3(NB * 32), 256, 0, stream>>>(x, P(gi), P(bi), y, H * W, lg2(C / 32));
  };
  // in16 != nullptr -> f16 input (async path); else f32 input via inF
  auto conv = [&](const void* in, bool in16, int wi, int bi, const float* res,
                  float* out, int H, int W, int Cin, int Cout, int Ks, int clip) {
    const int lW = lg2(W), lHW = lg2(H * W);
    const int lcb = (Cin % 32 == 0) ? lg2(Cin / 32) : 0;
    const float* bp = bi >= 0 ? P(bi) : nullptr;
    const _Float16* wh = WH(wi);
    const bool aF = (Cin % 32 == 0), nt2 = (Cout % 32 == 0);
    dim3 grid((NB * H * W / 16 + WAVES - 1) / WAVES,
              (Cout + (nt2 ? 31 : 15)) / (nt2 ? 32 : 16));
    if (in16 && aF && nt2) {
      if (Ks == 3)
        conv_wmma_async_kernel<3><<<grid, 256, 0, stream>>>(
            (const _Float16*)in, wh, bp, res, out, ZP, lW, lHW, Cin, Cout, lcb);
      else
        conv_wmma_async_kernel<1><<<grid, 256, 0, stream>>>(
            (const _Float16*)in, wh, bp, res, out, ZP, lW, lHW, Cin, Cout, lcb);
    } else if (Ks == 1) {            // nin 1x1: f32 input
      conv_wmma_sync_kernel<1, 2, true, true, false><<<grid, 256, 0, stream>>>(
          in, wh, bp, res, out, lW, lHW, Cin, Cout, lcb, clip);
    } else if (!aF) {                // conv_in: Cin = 3, f32 input
      conv_wmma_sync_kernel<3, 2, false, true, false><<<grid, 256, 0, stream>>>(
          in, wh, bp, res, out, lW, lHW, Cin, Cout, lcb, clip);
    } else {                         // final conv: Cout = 3, f16 input
      conv_wmma_sync_kernel<3, 1, true, false, true><<<grid, 256, 0, stream>>>(
          in, wh, bp, res, out, lW, lHW, Cin, Cout, lcb, clip);
    }
  };
  auto elemgrid = [&](size_t total) {
    size_t b = (total + 255) / 256; if (b > 65535) b = 65535; return (unsigned)b;
  };
  auto pool = [&](const float* in, float* out, int H, int W, int C) {
    pool2_kernel<<<elemgrid((size_t)NB * (H / 2) * (W / 2) * C), 256, 0, stream>>>(
        in, out, lg2(H / 2), lg2(W / 2), lg2(C));
  };
  auto upsm = [&](const float* in, _Float16* out, int H, int W, int C) {
    up2_kernel<<<elemgrid((size_t)NB * 4 * H * W * C), 256, 0, stream>>>(
        in, out, lg2(H), lg2(W), lg2(C));
  };
  float* cur = B0; float* oth = B1;
  auto swapb = [&]() { float* tmp = cur; cur = oth; oth = tmp; };

  // Residual block: out = conv(...)+x  (or h2 + nin(h2), faithful to source)
  auto resblock = [&](const RB& p, int H, int W, int cin, int cout) {
    gn(cur, p.n1g, p.n1b, G16, H, W, cin);
    conv(G16, true, p.c1, -1, nullptr, U32, H, W, cin, cout, 3, 0);
    gn(U32, p.n2g, p.n2b, G16, H, W, cout);
    if (p.nin < 0) {
      conv(G16, true, p.c2, -1, cur, oth, H, W, cout, cout, 3, 0);
    } else {
      conv(G16, true, p.c2, -1, nullptr, U32, H, W, cout, cout, 3, 0);
      conv(U32, false, p.nin, -1, U32, oth, H, W, cout, cout, 1, 0);
    }
    swapb();
  };

  // =========================== Encoder =====================================
  conv(P(x_i), false, enc_convin, -1, nullptr, cur, 256, 256, 3, 128, 3, 0);
  {
    const int cins[5]  = {128, 128, 128, 256, 256};
    const int couts[5] = {128, 128, 256, 256, 512};
    int H = 256;
    for (int i = 0; i < 5; ++i) {
      resblock(down_b[i][0], H, H, cins[i],  couts[i]);
      resblock(down_b[i][1], H, H, couts[i], couts[i]);
      if (i != 4) { pool(cur, oth, H, H, couts[i]); swapb(); H >>= 1; }
    }
  }
  resblock(emid[0], 16, 16, 512, 512);
  resblock(emid[1], 16, 16, 512, 512);
  gn(cur, enc_ng, enc_nb, G16, 16, 16, 512);
  conv(G16, true, enc_cow, enc_cob, nullptr, oth, 16, 16, 512, 256, 1, 0);

  // =========================== Vector quantization =========================
  _Float16* ZQ = G16;   // reuse f16 temp for codebook entries
  vq_kernel<<<dim3(NB * 16 * 16), 256, 0, stream>>>(oth, P(emb_i), ZQ);

  // =========================== Decoder =====================================
  conv(ZQ, true, dec_ciw, dec_cib, nullptr, cur, 16, 16, 256, 512, 3, 0);
  resblock(dmid[0], 16, 16, 512, 512);
  resblock(dmid[1], 16, 16, 512, 512);
  {
    const int upins[5]  = {128, 256, 256, 512, 512};
    const int upouts[5] = {128, 128, 256, 256, 512};
    int H = 16;
    for (int i = 4; i >= 0; --i) {
      resblock(up_b[i][0], H, H, upins[i],  upouts[i]);
      resblock(up_b[i][1], H, H, upouts[i], upouts[i]);
      if (i != 0) {
        upsm(cur, G16, H, H, upouts[i]); H <<= 1;
        conv(G16, true, up_uw[i], up_ub[i], nullptr, oth, H, H, upouts[i], upouts[i], 3, 0);
        swapb();
      }
    }
  }
  gn(cur, dec_ng, dec_nb, G16, 256, 256, 128);
  conv(G16, true, dec_cow, dec_cob, nullptr, (float*)d_out, 256, 256, 128, 3, 3, 1);
}
